// DANetAttention_87316685127758
// MI455X (gfx1250) — matrix operand order repaired and checked by reading, not compiled, not run on hardware
//
#include <hip/hip_runtime.h>
#include <hip/hip_bf16.h>

// ---------------------------------------------------------------------------
// DANet dual attention for MI455X (gfx1250). bf16 WMMA everywhere, flash-style
// fused PAM, implicit-GEMM conv3x3. All operand layouts are K-innermost so
// every WMMA fragment is loaded as 2x16-byte contiguous loads per lane
// (global_load_b128 / ds_load_b128) instead of 32x global_load_u16.
// ---------------------------------------------------------------------------

typedef __bf16 bf16;
typedef bf16  bf16x16 __attribute__((ext_vector_type(16)));
typedef bf16  bf16x8  __attribute__((ext_vector_type(8)));   // 16 bytes
typedef float floatx8 __attribute__((ext_vector_type(8)));

#define BATCH 4
#define CCH   256
#define RCH   32
#define NPIX  4096   // 64*64
#define HDIM  64
#define WDIM  64

__device__ inline bf16x16 make_frag(const bf16x8 lo, const bf16x8 hi) {
  bf16x16 r;
#pragma unroll
  for (int i = 0; i < 8; ++i) { r[i] = lo[i]; r[8 + i] = hi[i]; }
  return r;
}

// A fragment: 16(M) x 32(K), row-major, elem (m,k) at p[m*ld + k].
// Lane(m,kb) needs K=[kb,kb+16) (contiguous order, matching the B loaders):
// two contiguous 16B loads. Requires p 16B-aligned, ld % 8 == 0.
__device__ inline bf16x16 load_a_frag_vec(const bf16* p, int ld) {
  const int lane = threadIdx.x & 31;
  const int m  = lane & 15;
  const int kb = (lane & 16) ? 16 : 0;
  const bf16* q = p + (size_t)m * ld + kb;
  return make_frag(*(const bf16x8*)(q), *(const bf16x8*)(q + 8));
}

// B fragment: 32(K) x 16(N), sourced from an N-major (N x K row-major) array:
// elem (k,n) at p[n*ld + k]. Lane(n,kb) needs K=[kb,kb+16): two 16B loads.
__device__ inline bf16x16 load_bT_frag_vec(const bf16* p, int ld) {
  const int lane = threadIdx.x & 31;
  const int n  = lane & 15;
  const int kb = (lane & 16) ? 16 : 0;
  const bf16* q = p + (size_t)n * ld + kb;
  return make_frag(*(const bf16x8*)(q), *(const bf16x8*)(q + 8));
}

// Same, but zero the lane's column when (colbase + n) is outside [0, WDIM)
// (conv zero padding).
__device__ inline bf16x16 load_bT_frag_vec_guard(const bf16* p, int ld, int colbase) {
  const int lane = threadIdx.x & 31;
  const int n = lane & 15;
  if ((unsigned)(colbase + n) >= (unsigned)WDIM) { bf16x16 z = {}; return z; }
  const int kb = (lane & 16) ? 16 : 0;
  const bf16* q = p + (size_t)n * ld + kb;
  return make_frag(*(const bf16x8*)(q), *(const bf16x8*)(q + 8));
}

#define WMMA_BF16(A, B, ACC) \
  __builtin_amdgcn_wmma_f32_16x16x32_bf16(false, (A), false, (B), (short)0, (ACC), false, false)

// ---------------------------------------------------------------------------
// fp32 -> bf16 convert (same index)
__global__ void cvt_bf16_kernel(const float* __restrict__ in, bf16* __restrict__ out, int n) {
  int i = blockIdx.x * blockDim.x + threadIdx.x;
  if (i < n) out[i] = (bf16)in[i];
}

// x_t[b][n][c] = x[b][c][n]   (flat over B*N*C)
__global__ void transpose_x_kernel(const float* __restrict__ x, bf16* __restrict__ x_t) {
  const size_t i = (size_t)blockIdx.x * 256 + threadIdx.x;
  const int c = (int)(i & (CCH - 1));
  const int n = (int)((i >> 8) & (NPIX - 1));
  const int b = (int)(i >> 20);
  x_t[i] = (bf16)x[((size_t)b * CCH + c) * NPIX + n];
}

// fw_r[o][pos][i] = fuse_w[o][i][pos]   (flat over 256*9*512)
__global__ void reorder_fw_kernel(const float* __restrict__ fw, bf16* __restrict__ fw_r) {
  const size_t idx = (size_t)blockIdx.x * 256 + threadIdx.x;
  const int i   = (int)(idx & 511);
  const int pos = (int)((idx >> 9) % 9);
  const int o   = (int)(idx / (9 * 512));
  fw_r[idx] = (bf16)fw[(size_t)o * (2 * CCH * 9) + (size_t)i * 9 + pos];
}

// ---------------------------------------------------------------------------
// q/k/v 1x1-conv projections. One wave per 16x16 output tile.
// grid = B * 256(n-tiles) * 20(o-tiles): ot 0-1 -> q, 2-3 -> k, 4-19 -> v
__global__ void qkv_proj_kernel(const bf16* __restrict__ x_t,   // [B][N][C]
                                const bf16* __restrict__ qw,    // [32][256]
                                const bf16* __restrict__ kw,    // [32][256]
                                const bf16* __restrict__ vw,    // [256][256]
                                const float* __restrict__ qb,
                                const float* __restrict__ kbias,
                                const float* __restrict__ vb,
                                bf16* __restrict__ q_t,   // [B][N][32]
                                bf16* __restrict__ k_t,   // [B][N][32]
                                bf16* __restrict__ v_cm)  // [B][C][N]
{
  const int wg = blockIdx.x;
  const int ot = wg % 20;
  const int nt = (wg / 20) % 256;
  const int b  = wg / (20 * 256);
  const int n0 = nt * 16;
  const bf16* W; const float* bias; int o0;
  if (ot < 2)      { W = qw; bias = qb;    o0 = ot * 16; }
  else if (ot < 4) { W = kw; bias = kbias; o0 = (ot - 2) * 16; }
  else             { W = vw; bias = vb;    o0 = (ot - 4) * 16; }
  const bf16* xt = x_t + ((size_t)b * NPIX + n0) * CCH;   // [16 n][256 c]
  floatx8 acc = {};
#pragma unroll
  for (int kk = 0; kk < CCH; kk += 32) {
    bf16x16 a  = load_a_frag_vec(W + o0 * CCH + kk, CCH);
    bf16x16 bb = load_bT_frag_vec(xt + kk, CCH);
    acc = WMMA_BF16(a, bb, acc);
  }
  const int lane = threadIdx.x & 31;
  const int n  = n0 + (lane & 15);
  const int mh = (lane & 16) ? 8 : 0;
#pragma unroll
  for (int r = 0; r < 8; ++r) {
    const int o = o0 + r + mh;
    const float val = acc[r] + bias[o];
    if (ot < 2)      q_t[((size_t)b * NPIX + n) * RCH + o]  = (bf16)val;
    else if (ot < 4) k_t[((size_t)b * NPIX + n) * RCH + o]  = (bf16)val;
    else             v_cm[((size_t)b * CCH + o) * NPIX + n] = (bf16)val;
  }
}

// ---------------------------------------------------------------------------
// Fused PAM (flash-attention style). One block per (batch, 16-row tile of N).
#define PAM_WAVES 8
__global__ void __launch_bounds__(256)
pam_kernel(const float* __restrict__ x,       // [B][C][N] fp32
           const bf16*  __restrict__ q_t,     // [B][N][32]
           const bf16*  __restrict__ k_t,     // [B][N][32]
           const bf16*  __restrict__ v_cm,    // [B][C][N]
           const float* __restrict__ gamma_p,
           bf16* __restrict__ feats_t)        // [B][N][512] -> channels [0,256)
{
  __shared__ float s_stats[PAM_WAVES][16][2];
  __shared__ float s_ms[16][2];                          // per-row: max, 1/denom
  __shared__ __align__(16) bf16 s_p[PAM_WAVES][16][32];  // p tile [n][m] per wave
  __shared__ float s_out[CCH][16];

  const int tid  = threadIdx.x;
  const int wave = tid >> 5;
  const int lane = tid & 31;
  const int b  = blockIdx.x >> 8;
  const int nt = blockIdx.x & 255;
  const int n0 = nt * 16;
  const float gp = gamma_p[0];

  const bf16* qp = q_t + ((size_t)b * NPIX + n0) * RCH;   // [16][32]
  const bf16* kp = k_t + (size_t)b * NPIX * RCH;          // [4096][32]
  const bf16* vp = v_cm + (size_t)b * CCH * NPIX;

  const bf16x16 aq = load_a_frag_vec(qp, RCH);

  for (int i = tid; i < CCH * 16; i += 256) (&s_out[0][0])[i] = 0.f;

  // ---- pass 1: online row max / sum(exp) over m ----
  float rmax[8], rsum[8];
#pragma unroll
  for (int r = 0; r < 8; ++r) { rmax[r] = -3.0e38f; rsum[r] = 0.f; }

  for (int m0 = wave * 16; m0 < NPIX; m0 += PAM_WAVES * 16) {
    __builtin_prefetch(kp + (size_t)(m0 + PAM_WAVES * 16) * RCH, 0, 1);
    bf16x16 bk = load_bT_frag_vec(kp + (size_t)m0 * RCH, RCH);
    floatx8 e = {};
    e = WMMA_BF16(aq, bk, e);
#pragma unroll
    for (int r = 0; r < 8; ++r) {
      const float ev = e[r];
      if (ev > rmax[r]) { rsum[r] *= __expf(rmax[r] - ev); rmax[r] = ev; }
      rsum[r] += __expf(ev - rmax[r]);
    }
  }
  // combine across the 16 lanes that share each row (xor masks stay in half-wave)
#pragma unroll
  for (int s = 1; s < 16; s <<= 1) {
#pragma unroll
    for (int r = 0; r < 8; ++r) {
      const float om = __shfl_xor(rmax[r], s, 32);
      const float os = __shfl_xor(rsum[r], s, 32);
      const float nm = fmaxf(rmax[r], om);
      rsum[r] = rsum[r] * __expf(rmax[r] - nm) + os * __expf(om - nm);
      rmax[r] = nm;
    }
  }
  if (lane == 0 || lane == 16) {
    const int rh = (lane >> 1) & 8;
#pragma unroll
    for (int r = 0; r < 8; ++r) {
      s_stats[wave][rh + r][0] = rmax[r];
      s_stats[wave][rh + r][1] = rsum[r];
    }
  }
  __syncthreads();
  if (tid < 16) {
    float gm = -3.0e38f, gs = 0.f;
    for (int wv = 0; wv < PAM_WAVES; ++wv) {
      const float om = s_stats[wv][tid][0], os = s_stats[wv][tid][1];
      const float nm = fmaxf(gm, om);
      gs = gs * __expf(gm - nm) + os * __expf(om - nm);
      gm = nm;
    }
    s_ms[tid][0] = gm;
    s_ms[tid][1] = 1.0f / gs;
  }
  __syncthreads();

  const int mh = (lane & 16) ? 8 : 0;
  float rowM[8];
#pragma unroll
  for (int r = 0; r < 8; ++r) rowM[r] = s_ms[r + mh][0];

  floatx8 acc[16];
#pragma unroll
  for (int ct = 0; ct < 16; ++ct) acc[ct] = (floatx8){};

  // ---- pass 2: recompute energies, p = exp(e - M), out += V * p^T ----
  // s_p holds the tile in natural [n][m] orientation; reading it as the
  // B operand (K = m) is exactly the N-major layout load_bT_frag_vec wants.
  for (int m0 = wave * 32; m0 < NPIX; m0 += PAM_WAVES * 32) {
#pragma unroll
    for (int half = 0; half < 2; ++half) {
      bf16x16 bk = load_bT_frag_vec(kp + (size_t)(m0 + half * 16) * RCH, RCH);
      floatx8 e = {};
      e = WMMA_BF16(aq, bk, e);
      const int ml = half * 16 + (lane & 15);
#pragma unroll
      for (int r = 0; r < 8; ++r)
        s_p[wave][r + mh][ml] = (bf16)__expf(e[r] - rowM[r]);
    }
    __syncthreads();
    const bf16x16 bp = load_bT_frag_vec(&s_p[wave][0][0], 32);
#pragma unroll
    for (int ct = 0; ct < 16; ++ct) {
      bf16x16 av = load_a_frag_vec(vp + (size_t)(ct * 16) * NPIX + m0, NPIX);
      acc[ct] = WMMA_BF16(av, bp, acc[ct]);
    }
    __syncthreads();
  }

  // reduce partial outputs across waves via LDS atomics
  const int nn = lane & 15;
#pragma unroll
  for (int ct = 0; ct < 16; ++ct) {
#pragma unroll
    for (int r = 0; r < 8; ++r)
      atomicAdd(&s_out[ct * 16 + r + mh][nn], acc[ct][r]);
  }
  __syncthreads();

  // store position branch into channel-innermost feats_t (coalesced over c)
  for (int i = tid; i < 16 * CCH; i += 256) {
    const int n = i >> 8;
    const int c = i & (CCH - 1);
    const float pam = s_out[c][n] * s_ms[n][1];
    const float xin = x[((size_t)b * CCH + c) * NPIX + n0 + n];
    feats_t[((size_t)b * NPIX + n0 + n) * (2 * CCH) + c] = (bf16)(gp * pam + xin);
  }
}

// ---------------------------------------------------------------------------
// CAM energy  energy[b][c][d] = sum_n x[c][n]*x[d][n]. grid = B*16*16, 1 wave
__global__ void cam_energy_kernel(const bf16* __restrict__ xbf, float* __restrict__ energy) {
  const int wg = blockIdx.x;
  const int dt = wg & 15;
  const int ct = (wg >> 4) & 15;
  const int b  = wg >> 8;
  const bf16* xb = xbf + (size_t)b * CCH * NPIX;
  floatx8 acc = {};
  for (int k = 0; k < NPIX; k += 32) {
    bf16x16 a  = load_a_frag_vec(xb + (size_t)(ct * 16) * NPIX + k, NPIX);
    bf16x16 bb = load_bT_frag_vec(xb + (size_t)(dt * 16) * NPIX + k, NPIX);
    acc = WMMA_BF16(a, bb, acc);
  }
  const int lane = threadIdx.x & 31;
  const int n  = lane & 15;
  const int mh = (lane & 16) ? 8 : 0;
  float* eb = energy + (size_t)b * CCH * CCH;
#pragma unroll
  for (int r = 0; r < 8; ++r)
    eb[(size_t)(ct * 16 + r + mh) * CCH + dt * 16 + n] = acc[r];
}

// ---------------------------------------------------------------------------
// CAM softmax of (rowmax - e) over d. One block (256 threads) per row.
__global__ void __launch_bounds__(256)
cam_softmax_kernel(const float* __restrict__ energy, bf16* __restrict__ attn) {
  __shared__ float red[256];
  const int row = blockIdx.x;                 // b*256 + c
  const float e = energy[(size_t)row * CCH + threadIdx.x];
  red[threadIdx.x] = e; __syncthreads();
  for (int s = 128; s > 0; s >>= 1) {
    if (threadIdx.x < s) red[threadIdx.x] = fmaxf(red[threadIdx.x], red[threadIdx.x + s]);
    __syncthreads();
  }
  const float m1 = red[0]; __syncthreads();
  const float en = m1 - e;                    // energy_new
  red[threadIdx.x] = en; __syncthreads();
  for (int s = 128; s > 0; s >>= 1) {
    if (threadIdx.x < s) red[threadIdx.x] = fmaxf(red[threadIdx.x], red[threadIdx.x + s]);
    __syncthreads();
  }
  const float m2 = red[0]; __syncthreads();
  const float p = __expf(en - m2);
  red[threadIdx.x] = p; __syncthreads();
  for (int s = 128; s > 0; s >>= 1) {
    if (threadIdx.x < s) red[threadIdx.x] += red[threadIdx.x + s];
    __syncthreads();
  }
  attn[(size_t)row * CCH + threadIdx.x] = (bf16)(p / red[0]);
}

// ---------------------------------------------------------------------------
// CAM output + channel branch -> feats_t channels [256,512). grid = B*16*256
__global__ void cam_out_kernel(const bf16* __restrict__ attn,  // [B][256][256]
                               const bf16* __restrict__ x_t,   // [B][N][C]
                               const float* __restrict__ x,
                               const float* __restrict__ gamma_c,
                               bf16* __restrict__ feats_t) {
  const int wg = blockIdx.x;
  const int nt = wg & 255;
  const int ct = (wg >> 8) & 15;
  const int b  = wg >> 12;
  const bf16* ab = attn + (size_t)b * CCH * CCH + (size_t)(ct * 16) * CCH;
  const int n0 = nt * 16;
  const bf16* xt = x_t + ((size_t)b * NPIX + n0) * CCH;    // [16 n][256 d]
  floatx8 acc = {};
#pragma unroll
  for (int k = 0; k < CCH; k += 32) {
    bf16x16 a  = load_a_frag_vec(ab + k, CCH);
    bf16x16 bb = load_bT_frag_vec(xt + k, CCH);
    acc = WMMA_BF16(a, bb, acc);
  }
  const float gc = gamma_c[0];
  const int lane = threadIdx.x & 31;
  const int n  = n0 + (lane & 15);
  const int mh = (lane & 16) ? 8 : 0;
#pragma unroll
  for (int r = 0; r < 8; ++r) {
    const int c = ct * 16 + r + mh;
    const float val = gc * acc[r] + x[((size_t)b * CCH + c) * NPIX + n];
    feats_t[((size_t)b * NPIX + n) * (2 * CCH) + CCH + c] = (bf16)val;
  }
}

// ---------------------------------------------------------------------------
// conv3x3 (pad 1, no bias) as implicit GEMM: 9 shifted K=512 GEMMs.
// grid = B * 16(o-tiles) * 256(pixel tiles: 64 rows x 4 col-tiles), 1 wave
__global__ void conv_kernel(const bf16* __restrict__ feats_t, // [B][4096][512]
                            const bf16* __restrict__ fw_r,    // [256][9][512]
                            float* __restrict__ y)            // [B][256][4096]
{
  const int wg = blockIdx.x;
  const int nt = wg & 255;
  const int ot = (wg >> 8) & 15;
  const int b  = wg >> 12;
  const int h  = nt >> 2;
  const int w0 = (nt & 3) << 4;
  const bf16* ft = feats_t + (size_t)b * NPIX * (2 * CCH);
  floatx8 acc = {};
#pragma unroll 1
  for (int ky = 0; ky < 3; ++ky) {
    const int hh = h + ky - 1;
    if (hh < 0 || hh >= HDIM) continue;
#pragma unroll 1
    for (int kx = 0; kx < 3; ++kx) {
      const int ww = w0 + kx - 1;
      const int pos = ky * 3 + kx;
      const bf16* wb = fw_r + (size_t)(ot * 16) * (9 * 512) + (size_t)pos * 512;
      const bf16* fb = ft + (size_t)(hh * WDIM + ww) * (2 * CCH);
#pragma unroll 4
      for (int i0 = 0; i0 < 2 * CCH; i0 += 32) {
        bf16x16 a  = load_a_frag_vec(wb + i0, 9 * 512);
        bf16x16 bb = load_bT_frag_vec_guard(fb + i0, 2 * CCH, ww);
        acc = WMMA_BF16(a, bb, acc);
      }
    }
  }
  const int lane = threadIdx.x & 31;
  const int n  = w0 + (lane & 15);
  const int mh = (lane & 16) ? 8 : 0;
#pragma unroll
  for (int r = 0; r < 8; ++r) {
    const int c = ot * 16 + r + mh;
    y[((size_t)b * CCH + c) * NPIX + h * WDIM + n] = acc[r];
  }
}

// ---------------------------------------------------------------------------
// BN batch stats per channel. grid = 256 channels
__global__ void __launch_bounds__(256)
bn_stats_kernel(const float* __restrict__ y, float* __restrict__ stats) {
  __shared__ float ssum[256], ssq[256];
  const int c = blockIdx.x;
  float s = 0.f, q = 0.f;
  for (int b = 0; b < BATCH; ++b) {
    const float* yp = y + ((size_t)b * CCH + c) * NPIX;
    for (int i = threadIdx.x; i < NPIX; i += 256) { const float v = yp[i]; s += v; q += v * v; }
  }
  ssum[threadIdx.x] = s; ssq[threadIdx.x] = q; __syncthreads();
  for (int st = 128; st > 0; st >>= 1) {
    if (threadIdx.x < st) { ssum[threadIdx.x] += ssum[threadIdx.x + st]; ssq[threadIdx.x] += ssq[threadIdx.x + st]; }
    __syncthreads();
  }
  if (threadIdx.x == 0) {
    const float inv_n = 1.0f / (float)(BATCH * NPIX);
    const float mean = ssum[0] * inv_n;
    const float var  = ssq[0] * inv_n - mean * mean;
    stats[c]       = mean;
    stats[CCH + c] = rsqrtf(var + 1e-5f);
  }
}

// BN apply + affine + ReLU -> d_out
__global__ void bn_apply_kernel(const float* __restrict__ y, const float* __restrict__ stats,
                                const float* __restrict__ scale, const float* __restrict__ bias,
                                float* __restrict__ out) {
  const size_t i = (size_t)blockIdx.x * blockDim.x + threadIdx.x;
  const int c = (int)((i >> 12) & 255);
  const float v = (y[i] - stats[c]) * stats[CCH + c] * scale[c] + bias[c];
  out[i] = fmaxf(v, 0.f);
}

// ---------------------------------------------------------------------------
extern "C" void kernel_launch(void* const* d_in, const int* in_sizes, int n_in,
                              void* d_out, int out_size, void* d_ws, size_t ws_size,
                              hipStream_t stream) {
  const float* x       = (const float*)d_in[0];
  const float* q_w     = (const float*)d_in[1];
  const float* q_b     = (const float*)d_in[2];
  const float* k_w     = (const float*)d_in[3];
  const float* k_b     = (const float*)d_in[4];
  const float* v_w     = (const float*)d_in[5];
  const float* v_b     = (const float*)d_in[6];
  const float* gamma_p = (const float*)d_in[7];
  const float* gamma_c = (const float*)d_in[8];
  const float* fuse_w  = (const float*)d_in[9];
  const float* bn_sc   = (const float*)d_in[10];
  const float* bn_bi   = (const float*)d_in[11];
  float* out = (float*)d_out;

  char* base = (char*)d_ws;
  size_t off = 0;
  auto take = [&](size_t bytes) -> char* {
    char* p = base + off;
    off = (off + bytes + 255) & ~(size_t)255;
    return p;
  };
  bf16*  xbf     = (bf16*)take((size_t)BATCH * CCH * NPIX * 2);       // [B][C][N]
  bf16*  x_t     = (bf16*)take((size_t)BATCH * NPIX * CCH * 2);       // [B][N][C]
  bf16*  qw_bf   = (bf16*)take((size_t)RCH * CCH * 2);
  bf16*  kw_bf   = (bf16*)take((size_t)RCH * CCH * 2);
  bf16*  vw_bf   = (bf16*)take((size_t)CCH * CCH * 2);
  bf16*  fw_r    = (bf16*)take((size_t)CCH * 9 * 2 * CCH * 2);        // [O][9][512]
  bf16*  q_t     = (bf16*)take((size_t)BATCH * NPIX * RCH * 2);
  bf16*  k_t     = (bf16*)take((size_t)BATCH * NPIX * RCH * 2);
  bf16*  v_cm    = (bf16*)take((size_t)BATCH * CCH * NPIX * 2);
  bf16*  feats_t = (bf16*)take((size_t)BATCH * NPIX * 2 * CCH * 2);   // [B][N][512]
  float* energy  = (float*)take((size_t)BATCH * CCH * CCH * 4);
  bf16*  attn_c  = (bf16*)take((size_t)BATCH * CCH * CCH * 2);
  float* y       = (float*)take((size_t)BATCH * CCH * NPIX * 4);
  float* stats   = (float*)take((size_t)2 * CCH * 4);
  (void)ws_size; (void)n_in; (void)in_sizes; (void)out_size;

  // 1) conversions / layout permutes
  {
    int n = BATCH * CCH * NPIX;
    cvt_bf16_kernel<<<(n + 255) / 256, 256, 0, stream>>>(x, xbf, n);
    transpose_x_kernel<<<n / 256, 256, 0, stream>>>(x, x_t);
    n = RCH * CCH;
    cvt_bf16_kernel<<<(n + 255) / 256, 256, 0, stream>>>(q_w, qw_bf, n);
    cvt_bf16_kernel<<<(n + 255) / 256, 256, 0, stream>>>(k_w, kw_bf, n);
    n = CCH * CCH;
    cvt_bf16_kernel<<<(n + 255) / 256, 256, 0, stream>>>(v_w, vw_bf, n);
    reorder_fw_kernel<<<(CCH * 9 * 2 * CCH) / 256, 256, 0, stream>>>(fuse_w, fw_r);
  }
  // 2) q/k/v projections
  qkv_proj_kernel<<<BATCH * 256 * 20, 32, 0, stream>>>(
      x_t, qw_bf, kw_bf, vw_bf, q_b, k_b, v_b, q_t, k_t, v_cm);
  // 3) fused PAM -> feats_t[:, :, 0:256]
  pam_kernel<<<BATCH * 256, 256, 0, stream>>>(x, q_t, k_t, v_cm, gamma_p, feats_t);
  // 4) CAM
  cam_energy_kernel<<<BATCH * 16 * 16, 32, 0, stream>>>(xbf, energy);
  cam_softmax_kernel<<<BATCH * CCH, 256, 0, stream>>>(energy, attn_c);
  cam_out_kernel<<<BATCH * 16 * 256, 32, 0, stream>>>(attn_c, x_t, x, gamma_c, feats_t);
  // 5) fuse conv3x3 -> y
  conv_kernel<<<BATCH * 16 * 256, 32, 0, stream>>>(feats_t, fw_r, y);
  // 6) BN (batch stats) + ReLU -> out
  bn_stats_kernel<<<CCH, 256, 0, stream>>>(y, stats);
  bn_apply_kernel<<<(BATCH * CCH * NPIX) / 256, 256, 0, stream>>>(y, stats, bn_sc, bn_bi, out);
}
